// Decoder_11879879542455
// MI455X (gfx1250) — matrix operand order repaired and checked by reading, not compiled, not run on hardware
//
#include <hip/hip_runtime.h>

#define B_   128
#define L_   256
#define E_   256
#define H_   512
#define T_   256
#define BM   16      // batch rows per slab
#define GK   24      // gates K chunks: (E+H)/32 = 768/32
#define GWG  4       // workgroups cooperating on one batch slab
#define NGRP (B_/BM) // 8 slabs -> grid = 32

typedef float  v8f   __attribute__((ext_vector_type(8)));
typedef __bf16 v16bf __attribute__((ext_vector_type(16)));

union Frag {
  uint4 q[2];
  unsigned short s[16];
  v16bf v;
};

#define WMMA_BF16(A, Bf, C) \
  __builtin_amdgcn_wmma_f32_16x16x32_bf16(false, (A), false, (Bf), (short)0, (C), false, false)

__device__ __forceinline__ unsigned short f2bf(float f) {
  unsigned int u = __float_as_uint(f);
  u += 0x7FFFu + ((u >> 16) & 1u);          // RNE truncate to bf16
  return (unsigned short)(u >> 16);
}
__device__ __forceinline__ float sigmf(float x) { return 1.0f / (1.0f + __expf(-x)); }
__device__ __forceinline__ v8f zero8() { v8f z = {0.f,0.f,0.f,0.f,0.f,0.f,0.f,0.f}; return z; }

// Monotonic group barrier among GWG co-resident workgroups (persistent kernel).
// Counter zeroed by a prologue kernel each launch -> deterministic under graph replay.
__device__ __forceinline__ void group_barrier(unsigned int* cnt, unsigned int target) {
  __syncthreads();
  if (threadIdx.x == 0) {
    __threadfence();                        // release: flush our stores to device scope
    atomicAdd(cnt, 1u);
    while (__hip_atomic_load(cnt, __ATOMIC_ACQUIRE, __HIP_MEMORY_SCOPE_AGENT) < target)
      __builtin_amdgcn_s_sleep(1);
  }
  __syncthreads();
  __threadfence();                          // acquire: invalidate WGP$ before reading peers' data
}

static constexpr size_t OUT_PTR = (size_t)B_ * T_ * L_;          // alphas first
static constexpr size_t OUT_HF  = OUT_PTR + (size_t)B_ * T_;     // then pointers
static constexpr size_t OUT_CF  = OUT_HF  + (size_t)B_ * H_;     // then h_f, c_f

__global__ void zero_u32(unsigned int* p, int n) {
  int i = blockIdx.x * blockDim.x + threadIdx.x;
  if (i < n) p[i] = 0u;
}

// ---------------------------------------------------------------------------
// Pack all weights fp32 -> bf16 in WMMA B-fragment order:
// tile[(nt,kc)] holds 512 bf16; element e = lane*16 + j maps to
// (k = (lane<16?0:8)+(j&7)+(j>=8?16:0) + kc*32, n = (lane&15) + nt*16).
// ---------------------------------------------------------------------------
__global__ void pack_weights(const float* __restrict__ Wi2h, const float* __restrict__ Wh2h,
                             const float* __restrict__ Winp, const float* __restrict__ Wout,
                             const float* __restrict__ Wctx,
                             unsigned short* __restrict__ wgp, unsigned short* __restrict__ wip,
                             unsigned short* __restrict__ wop, unsigned short* __restrict__ wcp) {
  int idx = blockIdx.x * 256 + threadIdx.x;
  const int SZ_G = 128 * GK * 512;
  const int SZ_I = 32 * 16 * 512;
  const int SZ_O = 32 * 32 * 512;
  const int SZ_C = 32 * 16 * 512;
  int e = idx & 511;
  int lane = e >> 4, j = e & 15;
  int n = lane & 15;
  int k = ((lane >> 4) << 3) + (j & 7) + ((j >> 3) << 4); // 0..31 within K-chunk
  if (idx < SZ_G) {
    int tile = idx >> 9;
    int nt = tile / GK, kc = tile % GK;
    int row = nt * 16 + n;                        // gate output column
    float v = (kc < 8) ? Wi2h[row * E_ + kc * 32 + k]
                       : Wh2h[row * H_ + (kc - 8) * 32 + k];
    wgp[idx] = f2bf(v);
  } else if (idx < SZ_G + SZ_I) {
    int id2 = idx - SZ_G; int tile = id2 >> 9;
    int nt = tile >> 4, kc = tile & 15;
    wip[id2] = f2bf(Winp[(nt * 16 + n) * H_ + kc * 32 + k]);
  } else if (idx < SZ_G + SZ_I + SZ_O) {
    int id2 = idx - SZ_G - SZ_I; int tile = id2 >> 9;
    int nt = tile >> 5, kc = tile & 31;
    wop[id2] = f2bf(Wout[(nt * 16 + n) * (2 * H_) + kc * 32 + k]);
  } else if (idx < SZ_G + SZ_I + SZ_O + SZ_C) {
    int id2 = idx - SZ_G - SZ_I - SZ_O; int tile = id2 >> 9;
    int nt = tile >> 4, kc = tile & 15;
    wcp[id2] = f2bf(Wctx[(nt * 16 + n) * H_ + kc * 32 + k]);
  }
}

// ---------------------------------------------------------------------------
// One-time context projection: ctx[b][g][l] = W_ctx @ context[b,l,:] + b_ctx.
// GEMM [B*L, H] x [H, H] with bf16 WMMA, one wave per 16x16 output tile.
// ---------------------------------------------------------------------------
__global__ void ctx_gemm(const float* __restrict__ context, const unsigned short* __restrict__ wcp,
                         const float* __restrict__ bctx, float* __restrict__ ctx) {
  int gw   = (blockIdx.x * blockDim.x + threadIdx.x) >> 5;
  int lane = threadIdx.x & 31;
  int mt = gw >> 5;                               // 0..2047 row tiles (rows = b*L + l)
  int nt = gw & 31;                               // 0..31 col tiles
  int row = mt * 16 + (lane & 15);
  int kbF = (lane < 16) ? 0 : 8;                  // A-fragment half select
  const float* ap = context + (size_t)row * H_;
  v8f acc = zero8();
  for (int kc = 0; kc < 16; ++kc) {
    float4 f0 = *(const float4*)(ap + kc * 32 + kbF);
    float4 f1 = *(const float4*)(ap + kc * 32 + kbF + 4);
    float4 f2 = *(const float4*)(ap + kc * 32 + kbF + 16);
    float4 f3 = *(const float4*)(ap + kc * 32 + kbF + 20);
    float fv[16] = {f0.x,f0.y,f0.z,f0.w, f1.x,f1.y,f1.z,f1.w,
                    f2.x,f2.y,f2.z,f2.w, f3.x,f3.y,f3.z,f3.w};
    Frag a;
#pragma unroll
    for (int jj = 0; jj < 16; ++jj) a.s[jj] = f2bf(fv[jj]);
    Frag b;
    const uint4* bp = (const uint4*)(wcp + ((size_t)(nt * 16 + kc) << 9) + lane * 16);
    b.q[0] = bp[0]; b.q[1] = bp[1];
    acc = WMMA_BF16(a.v, b.v, acc);
  }
  int nn = lane & 15;
  int g = nt * 16 + nn;
  float bias = bctx[g];
  int mbase = mt * 16 + ((lane < 16) ? 0 : 8);
#pragma unroll
  for (int r = 0; r < 8; ++r) {
    int m = mbase + r;
    int b = m >> 8, l = m & 255;                  // m = b*L + l
    ctx[((size_t)b * H_ + g) * L_ + l] = acc[r] + bias;
  }
}

// ---------------------------------------------------------------------------
// Persistent decoder: 32 workgroups x 1024 threads. GWG=4 workgroups
// cooperate on one 16-row batch slab for all 256 steps. Cheap recurrent math
// (WMMA GEMMs, LSTM pointwise, softmax) is replicated in each WG (bitwise
// identical); the two expensive ctx scans (score tanh pass, hidden reduction)
// are split 4-way and exchanged through L2-resident scratch + group barriers.
// ---------------------------------------------------------------------------
__global__ void __launch_bounds__(1024)
ptrnet_persistent(const float* __restrict__ emb, const float* __restrict__ x0,
                  const float* __restrict__ h0,  const float* __restrict__ c0,
                  const float* __restrict__ bi2h, const float* __restrict__ bh2h,
                  const float* __restrict__ bout, const float* __restrict__ binp,
                  const float* __restrict__ Vv,
                  const unsigned short* __restrict__ wgp,
                  const unsigned short* __restrict__ wip,
                  const unsigned short* __restrict__ wop,
                  const float* __restrict__ ctx,
                  float* __restrict__ scw, float* __restrict__ hidw,
                  unsigned int* __restrict__ barcnt,
                  float* __restrict__ out) {
  extern __shared__ char smem_raw[];
  float* c_s    = (float*)smem_raw;                          // [BM][H]   carry c
  float* inp_s  = c_s   + BM * H_;                           // [BM][H]   attention proj
  float* sc_s   = inp_s + BM * H_;                           // [BM][L]   alpha (local copy)
  float* mask_s = sc_s  + BM * L_;                           // [BM][L]   pointer mask
  unsigned short* xh_s = (unsigned short*)(mask_s + BM * L_); // [BM][768] bf16 {x | h}
  unsigned short* hh_s = xh_s + BM * 768;                    // [BM][1024] bf16 {hidden | h_t}
  int* idx_s = (int*)(hh_s + BM * 1024);                     // [BM] argmax

  const int tid  = threadIdx.x;
  const int w    = tid >> 5;
  const int lane = tid & 31;
  const int blkg = blockIdx.x >> 2;                          // batch slab 0..7
  const int grp  = blockIdx.x & (GWG - 1);                   // wg-in-group 0..3
  const int bg0  = blkg * BM;
  const int nn   = lane & 15;
  const int mrow = (lane < 16) ? 0 : 8;                      // C/D fragment row base
  const int kbA  = (lane < 16) ? 0 : 8;                      // A fragment half select
  const int colb = w * 16 + nn;
  unsigned int* mycnt = barcnt + blkg;
  unsigned int bar_ep = 0;

  // per-thread loop-invariant biases (per-gate column colb)
  const float bI = bi2h[colb]         + bh2h[colb];
  const float bF = bi2h[512  + colb]  + bh2h[512  + colb];
  const float bG = bi2h[1024 + colb]  + bh2h[1024 + colb];
  const float bO = bi2h[1536 + colb]  + bh2h[1536 + colb];
  const float bInp = binp[colb];
  const float bOut = bout[colb];

  // --- init carries (every WG keeps an identical copy) ---
  for (int i = tid; i < BM * E_; i += 1024) {
    int m = i >> 8, e = i & 255;
    xh_s[m * 768 + e] = f2bf(x0[(size_t)(bg0 + m) * E_ + e]);
  }
  for (int i = tid; i < BM * H_; i += 1024) {
    int m = i >> 9, hc = i & 511;
    xh_s[m * 768 + 256 + hc] = f2bf(h0[(size_t)(bg0 + m) * H_ + hc]);
    c_s[i] = c0[(size_t)(bg0 + m) * H_ + hc];
  }
  for (int i = tid; i < BM * L_; i += 1024) mask_s[i] = 1.0f;
  __syncthreads();

  const unsigned short* arowg = xh_s + nn * 768;
  const unsigned short* arowo = hh_s + nn * 1024;

  for (int t = 0; t < T_; ++t) {
    // ========== phase G: gates GEMM (replicated; wave w owns i/f/g/o tiles) ==========
    v8f acc0 = zero8(), acc1 = zero8(), acc2 = zero8(), acc3 = zero8();
    for (int kc = 0; kc < GK; ++kc) {
      Frag a;
      const uint4* ap = (const uint4*)(arowg + kc * 32 + kbA);
      a.q[0] = ap[0]; a.q[1] = ap[2];                      // K lo/hi halves
      if (kc + 1 < GK)
        __builtin_prefetch(wgp + ((size_t)(w * GK + kc + 1) << 9) + lane * 16, 0, 0);
      Frag b;
      const uint4* p;
      p = (const uint4*)(wgp + ((size_t)((w      ) * GK + kc) << 9) + lane * 16);
      b.q[0] = p[0]; b.q[1] = p[1];  acc0 = WMMA_BF16(a.v, b.v, acc0);
      p = (const uint4*)(wgp + ((size_t)((32 + w ) * GK + kc) << 9) + lane * 16);
      b.q[0] = p[0]; b.q[1] = p[1];  acc1 = WMMA_BF16(a.v, b.v, acc1);
      p = (const uint4*)(wgp + ((size_t)((64 + w ) * GK + kc) << 9) + lane * 16);
      b.q[0] = p[0]; b.q[1] = p[1];  acc2 = WMMA_BF16(a.v, b.v, acc2);
      p = (const uint4*)(wgp + ((size_t)((96 + w ) * GK + kc) << 9) + lane * 16);
      b.q[0] = p[0]; b.q[1] = p[1];  acc3 = WMMA_BF16(a.v, b.v, acc3);
    }
    __syncthreads();                                       // all xh reads done

    // ========== LSTM pointwise, fully in registers (replicated) ==========
#pragma unroll
    for (int r = 0; r < 8; ++r) {
      int m = mrow + r;
      float iv = acc0[r] + bI, fv = acc1[r] + bF;
      float gv = acc2[r] + bG, ov = acc3[r] + bO;
      float cold = c_s[m * H_ + colb];
      float cn = sigmf(fv) * cold + sigmf(iv) * tanhf(gv);
      float ht = sigmf(ov) * tanhf(cn);
      c_s[m * H_ + colb] = cn;
      unsigned short hb = f2bf(ht);
      xh_s[m * 768  + 256 + colb] = hb;                    // for inp GEMM
      hh_s[m * 1024 + 512 + colb] = hb;                    // for out GEMM concat
    }
    __syncthreads();

    // ========== phase I: inp = h_t @ W_inp^T + b_inp (replicated) ==========
    v8f accI = zero8();
    for (int kc = 0; kc < 16; ++kc) {
      Frag a;
      const uint4* ap = (const uint4*)(arowg + 256 + kc * 32 + kbA);
      a.q[0] = ap[0]; a.q[1] = ap[2];
      Frag b;
      const uint4* p = (const uint4*)(wip + ((size_t)(w * 16 + kc) << 9) + lane * 16);
      b.q[0] = p[0]; b.q[1] = p[1];
      accI = WMMA_BF16(a.v, b.v, accI);
    }
#pragma unroll
    for (int r = 0; r < 8; ++r)
      inp_s[(mrow + r) * H_ + colb] = accI[r] + bInp;
    __syncthreads();

    // ========== phase S: SPLIT. This WG covers l in [grp*64, grp*64+64) ==========
    {
      int task = w;                                        // 32 tasks = 16 b x 2 l-blocks
      int b = task >> 1;
      int l = (grp << 6) + ((task & 1) << 5) + lane;
      const float* cp = ctx + (size_t)(bg0 + b) * H_ * L_ + l;
      const float* ip = inp_s + b * H_;
      float s = 0.f;
      for (int h = 0; h < H_; ++h)
        s = __builtin_fmaf(Vv[h], tanhf(ip[h] + cp[(size_t)h * L_]), s);
      scw[(size_t)blkg * BM * L_ + b * L_ + l] = s;
    }
    group_barrier(mycnt, (++bar_ep) * GWG);                // scores visible slab-wide

    // ========== phase M: masked softmax + argmax + mask update (replicated) ==========
    if (w < BM) {
      int b = w;
      const float* srow = scw + (size_t)blkg * BM * L_ + b * L_;
      float sv[8];
      float vmax = -__builtin_inff(); int amax = 0;
#pragma unroll
      for (int k8 = 0; k8 < 8; ++k8) {
        int l = lane + (k8 << 5);
        float mk = mask_s[b * L_ + l];
        float s = (mk == 0.f) ? -__builtin_inff() : srow[l];
        sv[k8] = s;
        if (s > vmax) { vmax = s; amax = l; }
      }
      for (int off = 16; off > 0; off >>= 1) {             // wave32 arg-max tree
        float ov = __shfl_xor(vmax, off, 32);
        int   oa = __shfl_xor(amax, off, 32);
        if (ov > vmax || (ov == vmax && oa < amax)) { vmax = ov; amax = oa; }
      }
      float sum = 0.f, ev[8];
#pragma unroll
      for (int k8 = 0; k8 < 8; ++k8) { ev[k8] = __expf(sv[k8] - vmax); sum += ev[k8]; }
      for (int off = 16; off > 0; off >>= 1) sum += __shfl_xor(sum, off, 32);
      float rs = 1.f / sum;
#pragma unroll
      for (int k8 = 0; k8 < 8; ++k8) {
        int l = lane + (k8 << 5);
        float al = ev[k8] * rs;
        sc_s[b * L_ + l] = al;
        if (grp == 0)
          out[((size_t)(bg0 + b) * T_ + t) * L_ + l] = al; // alphas output
      }
      if (lane == 0) {
        idx_s[b] = amax;
        mask_s[b * L_ + amax] = 0.f;
        if (grp == 0)
          out[OUT_PTR + (size_t)(bg0 + b) * T_ + t] = (float)amax;
      }
    }
    __syncthreads();

    // ========== phase H: SPLIT. This WG covers h in [grp*128, grp*128+128) ==========
    for (int rep = 0; rep < 8; ++rep) {
      int task = w + (rep << 5);                           // 256 tasks = 16 b x 16 h-blocks
      int b = task >> 4;
      int hb = task & 15;
      const float* cpb = ctx + (size_t)(bg0 + b) * H_ * L_;
      float av[8];
#pragma unroll
      for (int k8 = 0; k8 < 8; ++k8) av[k8] = sc_s[b * L_ + lane + (k8 << 5)];
#pragma unroll
      for (int h2 = 0; h2 < 8; ++h2) {
        int h = (grp << 7) + (hb << 3) + h2;
        const float* cr = cpb + (size_t)h * L_;
        float p = 0.f;
#pragma unroll
        for (int k8 = 0; k8 < 8; ++k8)
          p = __builtin_fmaf(av[k8], cr[lane + (k8 << 5)], p);
        for (int off = 16; off > 0; off >>= 1) p += __shfl_xor(p, off, 32);
        if (lane == 0) hidw[(size_t)blkg * BM * H_ + b * H_ + h] = p;
      }
    }
    group_barrier(mycnt, (++bar_ep) * GWG);                // hidden visible slab-wide

    // gather full hidden into local hh_s (bf16) for the out GEMM
    for (int i = tid; i < BM * H_; i += 1024) {
      int m = i >> 9, hc = i & 511;
      hh_s[m * 1024 + hc] = f2bf(hidw[(size_t)blkg * BM * H_ + i]);
    }
    __syncthreads();

    // ========== phase O: h_new = tanh(concat(hidden,h_t) @ W_out^T + b_out) ==========
    v8f accO = zero8();
    for (int kc = 0; kc < 32; ++kc) {
      Frag a;
      const uint4* ap = (const uint4*)(arowo + kc * 32 + kbA);
      a.q[0] = ap[0]; a.q[1] = ap[2];
      Frag b;
      const uint4* p = (const uint4*)(wop + ((size_t)(w * 32 + kc) << 9) + lane * 16);
      b.q[0] = p[0]; b.q[1] = p[1];
      accO = WMMA_BF16(a.v, b.v, accO);
    }
#pragma unroll
    for (int r = 0; r < 8; ++r) {
      int m = mrow + r;
      float hn = tanhf(accO[r] + bOut);
      xh_s[m * 768 + 256 + colb] = f2bf(hn);               // carried h for next step
      if (t == T_ - 1 && grp == 0)
        out[OUT_HF + (size_t)(bg0 + m) * H_ + colb] = hn;
    }

    // ========== phase X: gather next_x = embedded_inputs[b, argmax, :] ==========
    for (int i = tid; i < BM * E_; i += 1024) {
      int m = i >> 8, e = i & 255;
      int ix = idx_s[m];
      xh_s[m * 768 + e] = f2bf(emb[((size_t)(bg0 + m) * L_ + ix) * E_ + e]);
    }
    __syncthreads();
  }

  // final c_f
  if (grp == 0)
    for (int i = tid; i < BM * H_; i += 1024) {
      int m = i >> 9, hc = i & 511;
      out[OUT_CF + (size_t)(bg0 + m) * H_ + hc] = c_s[i];
    }
}

// ---------------------------------------------------------------------------
extern "C" void kernel_launch(void* const* d_in, const int* in_sizes, int n_in,
                              void* d_out, int out_size, void* d_ws, size_t ws_size,
                              hipStream_t stream) {
  (void)in_sizes; (void)n_in; (void)out_size; (void)ws_size;
  const float* emb  = (const float*)d_in[0];
  const float* x0   = (const float*)d_in[1];
  const float* h0   = (const float*)d_in[2];
  const float* c0   = (const float*)d_in[3];
  const float* context = (const float*)d_in[4];
  const float* Wi2h = (const float*)d_in[5];
  const float* bi2h = (const float*)d_in[6];
  const float* Wh2h = (const float*)d_in[7];
  const float* bh2h = (const float*)d_in[8];
  const float* Wout = (const float*)d_in[9];
  const float* bout = (const float*)d_in[10];
  const float* Winp = (const float*)d_in[11];
  const float* binp = (const float*)d_in[12];
  const float* Wctx = (const float*)d_in[13];
  const float* bctx = (const float*)d_in[14];
  const float* Vv   = (const float*)d_in[15];

  char* ws = (char*)d_ws;
  unsigned short* wgp = (unsigned short*)ws;  ws += (size_t)128 * GK * 512 * 2;
  unsigned short* wip = (unsigned short*)ws;  ws += (size_t)32 * 16 * 512 * 2;
  unsigned short* wop = (unsigned short*)ws;  ws += (size_t)32 * 32 * 512 * 2;
  unsigned short* wcp = (unsigned short*)ws;  ws += (size_t)32 * 16 * 512 * 2;
  float* ctx  = (float*)ws;                   ws += (size_t)B_ * H_ * L_ * 4;  // 64MB, L2-resident
  float* scw  = (float*)ws;                   ws += (size_t)NGRP * BM * L_ * 4;
  float* hidw = (float*)ws;                   ws += (size_t)NGRP * BM * H_ * 4;
  unsigned int* barcnt = (unsigned int*)ws;

  zero_u32<<<1, 64, 0, stream>>>(barcnt, NGRP);
  pack_weights<<<10240, 256, 0, stream>>>(Wi2h, Wh2h, Winp, Wout, Wctx, wgp, wip, wop, wcp);
  ctx_gemm<<<8192, 256, 0, stream>>>(context, wcp, bctx, ctx);

  constexpr int SMEM_MAIN =
      (BM * H_ * 2 + BM * L_ * 2) * 4 + (BM * 768 + BM * 1024) * 2 + BM * 4;
  (void)hipFuncSetAttribute((const void*)ptrnet_persistent,
                            hipFuncAttributeMaxDynamicSharedMemorySize, SMEM_MAIN);
  ptrnet_persistent<<<NGRP * GWG, 1024, SMEM_MAIN, stream>>>(
      emb, x0, h0, c0, bi2h, bh2h, bout, binp, Vv, wgp, wip, wop, ctx,
      scw, hidw, barcnt, (float*)d_out);
}
